// PositionwiseConvFFMoE_22385369547514
// MI455X (gfx1250) — matrix operand order repaired and checked by reading, not compiled, not run on hardware
//
#include <hip/hip_runtime.h>
#include <math.h>

// ---------------- problem constants (match reference) ----------------
constexpr int Bb = 4, Tt = 2048, Cc = 1024, Ff = 2048, Ee = 8;
constexpr int BT = Bb * Tt;          // 8192 tokens
constexpr int TILE_M = 32;           // tokens per workgroup tile (2 WMMA M-tiles)
constexpr int NTILES = BT / TILE_M;  // 256 workgroups
constexpr int FC = 256;              // F-chunk staged in LDS per iteration
constexpr int XSTR = Cc + 8;         // padded LDS stride (516 dw % 64 = 4 -> no bank conflicts)
constexpr int HSTR = FC + 8;         // padded LDS stride (132 dw % 64 = 4)

// d_out layout: concat(out[BT*C] f32, logits[BT*E] f32, probs[BT*E] f32, idx[BT*2] i32)
constexpr size_t OUT_LOG = (size_t)BT * Cc;
constexpr size_t OUT_PRB = OUT_LOG + (size_t)BT * Ee;
constexpr size_t OUT_IDX = OUT_PRB + (size_t)BT * Ee;

// workspace layout (bytes)
constexpr size_t WS_XBF = 0;                                  // BT*C bf16            = 16 MB
constexpr size_t WS_W1  = WS_XBF + (size_t)BT * Cc * 2;       // E*F*C bf16 [E][F][C] = 32 MB
constexpr size_t WS_W2  = WS_W1 + (size_t)Ee * Ff * Cc * 2;   // E*C*F bf16 [E][C][F] = 32 MB
constexpr size_t WS_CE  = WS_W2 + (size_t)Ee * Cc * Ff * 2;   // BT*E f32

typedef __attribute__((ext_vector_type(16))) __bf16 v16bf;
typedef __attribute__((ext_vector_type(8)))  __bf16 v8bf;
typedef __attribute__((ext_vector_type(8)))  float  v8f;

union ABFrag { v16bf v; v8bf h[2]; };

__device__ __forceinline__ float gelu_tanh(float v) {
    const float k = 0.7978845608028654f;
    float u = k * (v + 0.044715f * v * v * v);
    return 0.5f * v * (1.0f + tanhf(u));
}

// ---------------- f32 -> bf16 conversion / transposition ----------------
__global__ void cvt_x_kernel(const float* __restrict__ x, __bf16* __restrict__ o, int n) {
    int i = blockIdx.x * blockDim.x + threadIdx.x;
    if (i < n) o[i] = (__bf16)x[i];
}

// W1 (E,C,F) -> w1bf [E][F][C]
__global__ void cvt_w1_kernel(const float* __restrict__ w, __bf16* __restrict__ o) {
    int i = blockIdx.x * blockDim.x + threadIdx.x;   // over E*F*C = 16.8M
    int e = i / (Ff * Cc);
    int rem = i - e * (Ff * Cc);
    int f = rem / Cc;
    int c = rem - f * Cc;
    o[i] = (__bf16)w[((size_t)e * Cc + c) * Ff + f];
}

// W2 (E,F,C) -> w2bf [E][C][F]
__global__ void cvt_w2_kernel(const float* __restrict__ w, __bf16* __restrict__ o) {
    int i = blockIdx.x * blockDim.x + threadIdx.x;   // over E*C*F
    int e = i / (Cc * Ff);
    int rem = i - e * (Cc * Ff);
    int c = rem / Ff;
    int f = rem - c * Ff;
    o[i] = (__bf16)w[((size_t)e * Ff + f) * Cc + c];
}

// ---------------- router: logits, softmax, top-2, combine weights ----------------
__global__ void router_kernel(const float* __restrict__ x, const int* __restrict__ xm,
                              const float* __restrict__ Wr, float* __restrict__ out,
                              float* __restrict__ ceg) {
    int wave = threadIdx.x >> 5;
    int lane = threadIdx.x & 31;
    int t = blockIdx.x * 8 + wave;
    if (t >= BT) return;

    const float* xr = x + (size_t)t * Cc;
    float acc[Ee];
#pragma unroll
    for (int e = 0; e < Ee; ++e) acc[e] = 0.0f;
    for (int c = lane; c < Cc; c += 32) {
        float xv = xr[c];
        const float* wr = Wr + (size_t)c * Ee;
#pragma unroll
        for (int e = 0; e < Ee; ++e) acc[e] += xv * wr[e];
    }
#pragma unroll
    for (int e = 0; e < Ee; ++e) {
#pragma unroll
        for (int off = 16; off > 0; off >>= 1)
            acc[e] += __shfl_xor(acc[e], off, 32);
    }

    if (lane == 0) {
        int m = xm[t];
        float* lg = out + OUT_LOG + (size_t)t * Ee;
        float* pb = out + OUT_PRB + (size_t)t * Ee;
        int*   ix = (int*)(out + OUT_IDX) + (size_t)t * 2;
        float* ce = ceg + (size_t)t * Ee;
        if (m == 0) {
#pragma unroll
            for (int e = 0; e < Ee; ++e) { lg[e] = 0.0f; pb[e] = 0.0f; ce[e] = 0.0f; }
            ix[0] = -1; ix[1] = -1;
        } else {
            float mx = acc[0];
#pragma unroll
            for (int e = 1; e < Ee; ++e) mx = fmaxf(mx, acc[e]);
            float p[Ee], s = 0.0f;
#pragma unroll
            for (int e = 0; e < Ee; ++e) { p[e] = __expf(acc[e] - mx); s += p[e]; }
            float inv = 1.0f / s;
#pragma unroll
            for (int e = 0; e < Ee; ++e) p[e] *= inv;
            int i0 = 0;
#pragma unroll
            for (int e = 1; e < Ee; ++e) if (p[e] > p[i0]) i0 = e;
            int i1 = (i0 == 0) ? 1 : 0;
#pragma unroll
            for (int e = 0; e < Ee; ++e) if (e != i0 && p[e] > p[i1]) i1 = e;
            float wsum = p[i0] + p[i1];
            float winv = (wsum > 0.0f) ? (1.0f / wsum) : 1.0f;
#pragma unroll
            for (int e = 0; e < Ee; ++e) { lg[e] = acc[e]; pb[e] = p[e]; ce[e] = 0.0f; }
            ce[i0] = p[i0] * winv;
            ce[i1] = p[i1] * winv;
            ix[0] = i0; ix[1] = i1;
        }
    }
}

// ---------------- fused MoE FFN: out = sum_e ce_e * gelu(x@W1_e) @ W2_e ----------------
__global__ __launch_bounds__(256)
void moe_ffn_kernel(const __bf16* __restrict__ xbf,
                    const __bf16* __restrict__ w1bf,   // [E][F][C]
                    const __bf16* __restrict__ w2bf,   // [E][C][F]
                    const float*  __restrict__ ceg,    // [BT][E]
                    float* __restrict__ out) {
    __shared__ __attribute__((aligned(16))) __bf16 xsh[TILE_M][XSTR]; // ~66 KB, x tile
    __shared__ __attribute__((aligned(16))) __bf16 hsh[TILE_M][HSTR]; // ~17 KB, gelu(xW1)*ce
    __shared__ float sce[TILE_M][Ee];
    __shared__ int   sflag[Ee];

    const int tid  = threadIdx.x;
    const int wave = tid >> 5;     // 0..7
    const int lane = tid & 31;
    const int half = lane >> 4;    // 0/1
    const int l16  = lane & 15;
    const int tbase = blockIdx.x * TILE_M;

    // stage combine weights + x tile (bf16) into LDS
    sce[tid >> 3][tid & 7] = ceg[(size_t)(tbase + (tid >> 3)) * Ee + (tid & 7)];
    for (int j = tid; j < TILE_M * (Cc / 8); j += 256) {     // 8 bf16 (16B) units
        int row = j >> 7;            // Cc/8 = 128 units per row
        int col = (j & 127) * 8;
        *(v8bf*)(&xsh[row][col]) = *(const v8bf*)(xbf + (size_t)(tbase + row) * Cc + col);
    }
    __syncthreads();
    if (tid < Ee) {
        int f = 0;
        for (int r = 0; r < TILE_M; ++r) f |= (sce[r][tid] != 0.0f) ? 1 : 0;
        sflag[tid] = f;
    }

    // output accumulators: wave owns columns [wave*128, wave*128+128), 2 M-tiles
    v8f acc[16];
#pragma unroll
    for (int n = 0; n < 16; ++n) acc[n] = v8f{};

    for (int e = 0; e < Ee; ++e) {
        __syncthreads();                 // sflag visible / hsh reuse safe
        if (!sflag[e]) continue;         // uniform branch: EXEC stays all-ones
        const __bf16* w1e = w1bf + (size_t)e * Ff * Cc;
        const __bf16* w2e = w2bf + (size_t)e * Cc * Ff;

        for (int fc = 0; fc < Ff; fc += FC) {
            // ---- GEMM1: h[32][FC] = gelu(x_tile @ W1_e[:, fc:fc+FC]) * ce_row ----
            v8f h00 = v8f{}, h01 = v8f{}, h10 = v8f{}, h11 = v8f{};
            const int nt0 = 2 * wave, nt1 = 2 * wave + 1;
            const __bf16* b0p = w1e + (size_t)(fc + nt0 * 16 + l16) * Cc;
            const __bf16* b1p = w1e + (size_t)(fc + nt1 * 16 + l16) * Cc;
            // 2-deep software pipeline over K: next K-step's B-fragments are in
            // flight while the current K-step's 4 WMMAs execute.
            ABFrag bc0, bc1;
            bc0.h[0] = *(const v8bf*)(b0p + half * 16);
            bc0.h[1] = *(const v8bf*)(b0p + half * 16 + 8);
            bc1.h[0] = *(const v8bf*)(b1p + half * 16);
            bc1.h[1] = *(const v8bf*)(b1p + half * 16 + 8);
            for (int k = 0; k < Cc; k += 32) {
                ABFrag bn0, bn1;
                if (k + 32 < Cc) {
                    bn0.h[0] = *(const v8bf*)(b0p + k + 32 + half * 16);
                    bn0.h[1] = *(const v8bf*)(b0p + k + 32 + half * 16 + 8);
                    bn1.h[0] = *(const v8bf*)(b1p + k + 32 + half * 16);
                    bn1.h[1] = *(const v8bf*)(b1p + k + 32 + half * 16 + 8);
                }
                ABFrag a0, a1;
                a0.h[0] = *(const v8bf*)(&xsh[l16][k + half * 16]);
                a0.h[1] = *(const v8bf*)(&xsh[l16][k + half * 16 + 8]);
                a1.h[0] = *(const v8bf*)(&xsh[16 + l16][k + half * 16]);
                a1.h[1] = *(const v8bf*)(&xsh[16 + l16][k + half * 16 + 8]);
                h00 = __builtin_amdgcn_wmma_f32_16x16x32_bf16(false, a0.v, false, bc0.v,
                                                              (short)0, h00, false, false);
                h01 = __builtin_amdgcn_wmma_f32_16x16x32_bf16(false, a0.v, false, bc1.v,
                                                              (short)0, h01, false, false);
                h10 = __builtin_amdgcn_wmma_f32_16x16x32_bf16(false, a1.v, false, bc0.v,
                                                              (short)0, h10, false, false);
                h11 = __builtin_amdgcn_wmma_f32_16x16x32_bf16(false, a1.v, false, bc1.v,
                                                              (short)0, h11, false, false);
                bc0 = bn0;
                bc1 = bn1;
            }
            // epilogue: gelu + per-token ce scale (row scaling commutes through GEMM2)
#pragma unroll
            for (int r = 0; r < 8; ++r) {
                int m0 = r + 8 * half;
                int m1 = 16 + m0;
                float cs0 = sce[m0][e];
                float cs1 = sce[m1][e];
                hsh[m0][nt0 * 16 + l16] = (__bf16)(cs0 * gelu_tanh(h00[r]));
                hsh[m0][nt1 * 16 + l16] = (__bf16)(cs0 * gelu_tanh(h01[r]));
                hsh[m1][nt0 * 16 + l16] = (__bf16)(cs1 * gelu_tanh(h10[r]));
                hsh[m1][nt1 * 16 + l16] = (__bf16)(cs1 * gelu_tanh(h11[r]));
            }
            __syncthreads();

            // ---- GEMM2: acc[32 x 128] += h[32][FC] @ W2_e[fc:fc+FC, wave cols] ----
            for (int ks = 0; ks < FC; ks += 32) {
                ABFrag a0, a1;
                a0.h[0] = *(const v8bf*)(&hsh[l16][ks + half * 16]);
                a0.h[1] = *(const v8bf*)(&hsh[l16][ks + half * 16 + 8]);
                a1.h[0] = *(const v8bf*)(&hsh[16 + l16][ks + half * 16]);
                a1.h[1] = *(const v8bf*)(&hsh[16 + l16][ks + half * 16 + 8]);
                // 2-deep software pipeline over N-tiles: fragment nt+1 is in
                // flight while the 2 WMMAs of fragment nt execute.
                const __bf16* bp = w2e + (size_t)(wave * 128 + l16) * Ff + fc + ks;
                ABFrag bcur, bnext;
                bcur.h[0] = *(const v8bf*)(bp + half * 16);
                bcur.h[1] = *(const v8bf*)(bp + half * 16 + 8);
#pragma unroll
                for (int nt = 0; nt < 8; ++nt) {
                    if (nt < 7) {
                        const __bf16* bq = bp + (size_t)(nt + 1) * 16 * Ff;
                        bnext.h[0] = *(const v8bf*)(bq + half * 16);
                        bnext.h[1] = *(const v8bf*)(bq + half * 16 + 8);
                    }
                    acc[nt] = __builtin_amdgcn_wmma_f32_16x16x32_bf16(false, a0.v, false, bcur.v,
                                                                      (short)0, acc[nt], false, false);
                    acc[8 + nt] = __builtin_amdgcn_wmma_f32_16x16x32_bf16(false, a1.v, false, bcur.v,
                                                                          (short)0, acc[8 + nt], false, false);
                    bcur = bnext;
                }
            }
            __syncthreads();   // before next fchunk overwrites hsh
        }
    }

    // store out tile (f32)
#pragma unroll
    for (int nt = 0; nt < 8; ++nt) {
        int c = wave * 128 + nt * 16 + l16;
#pragma unroll
        for (int r = 0; r < 8; ++r) {
            int m = r + 8 * half;
            out[(size_t)(tbase + m) * Cc + c] = acc[nt][r];
            out[(size_t)(tbase + 16 + m) * Cc + c] = acc[8 + nt][r];
        }
    }
}

// ---------------- launch ----------------
extern "C" void kernel_launch(void* const* d_in, const int* in_sizes, int n_in,
                              void* d_out, int out_size, void* d_ws, size_t ws_size,
                              hipStream_t stream) {
    (void)in_sizes; (void)n_in; (void)out_size; (void)ws_size;
    const float* x  = (const float*)d_in[0];
    const int*   xm = (const int*)d_in[1];
    const float* Wr = (const float*)d_in[2];
    const float* W1 = (const float*)d_in[3];
    const float* W2 = (const float*)d_in[4];
    float* out = (float*)d_out;
    char* ws = (char*)d_ws;
    __bf16* xbf  = (__bf16*)(ws + WS_XBF);
    __bf16* w1bf = (__bf16*)(ws + WS_W1);
    __bf16* w2bf = (__bf16*)(ws + WS_W2);
    float*  ceg  = (float*)(ws + WS_CE);

    {
        int n = BT * Cc;
        cvt_x_kernel<<<(n + 255) / 256, 256, 0, stream>>>(x, xbf, n);
    }
    {
        int n = Ee * Ff * Cc;
        cvt_w1_kernel<<<n / 256, 256, 0, stream>>>(W1, w1bf);
        cvt_w2_kernel<<<n / 256, 256, 0, stream>>>(W2, w2bf);
    }
    router_kernel<<<BT / 8, 256, 0, stream>>>(x, xm, Wr, out, ceg);
    moe_ffn_kernel<<<NTILES, 256, 0, stream>>>(xbf, w1bf, w2bf, ceg, out);
}
